// GCN_59974923322073
// MI455X (gfx1250) — compile-verified, matrix operand order checked by reading, not run on hardware
//
#include <hip/hip_runtime.h>
#include <hip/hip_bf16.h>
#include <math.h>

// CDNA5 (gfx1250) wave32 WMMA types
typedef __attribute__((ext_vector_type(16))) _Float16 v16h;
typedef __attribute__((ext_vector_type(8)))  _Float16 v8h;
typedef __attribute__((ext_vector_type(2)))  _Float16 v2h;
typedef __attribute__((ext_vector_type(8)))  float    v8f;

#define NPG 36    // nodes per graph
#define EPG 220   // directed edges per graph (fixed topology)
#define HS  16    // H row stride (floats); layers 0-2 have fout <= 16

// ---------------------------------------------------------------------------
// Prep: build the fixed dense GCN propagation operator A_hat[48x64] (padded).
// A_hat[d][s] = sum over edges (s->d) of inv[s]*inv[d];  A_hat[i][i] += inv[i]^2
// deg[i] = 1 + in-degree(i). Topology identical for all graphs -> compute once.
// ---------------------------------------------------------------------------
__global__ void gcn_prep(const int* __restrict__ src, const int* __restrict__ dst,
                         float* __restrict__ Aout /* 48*64 floats in ws */) {
  __shared__ float inv[NPG];
  __shared__ float Amat[NPG * NPG];
  const int t = threadIdx.x;
  for (int i = t; i < NPG * NPG; i += blockDim.x) Amat[i] = 0.f;
  __syncthreads();
  if (t < NPG) {
    float d = 1.f;
    for (int e = 0; e < EPG; ++e) d += (dst[e] == t) ? 1.f : 0.f;
    inv[t] = rsqrtf(d);
  }
  __syncthreads();
  if (t < NPG) {
    const float it = inv[t];
    for (int e = 0; e < EPG; ++e)
      if (dst[e] == t) Amat[t * NPG + src[e]] += it * inv[src[e]];
    Amat[t * NPG + t] += it * it;   // self-loop (snorm) folded into diagonal
  }
  __syncthreads();
  for (int i = t; i < 48 * 64; i += blockDim.x) {
    const int r = i >> 6, c = i & 63;
    Aout[i] = (r < NPG && c < NPG) ? Amat[r * NPG + c] : 0.f;
  }
}

// T = H*W. Each lane owns a fixed output feature column (lrow for FPAD=16,
// lane for FPAD=32; weight column pre-masked to zero beyond fout) and a
// contiguous node range, processing node PAIRS so each store is one packed
// 2xf16 ds_store_b32 into T[f*64 + n].
template <int FIN, int FPAD>
__device__ __forceinline__ void transform(const float* Hw, _Float16* Tw,
                                          const float (&Wc)[FIN],
                                          int lane, int lrow) {
  const int npairs = (FPAD == 32) ? 18 : 9;          // 36 or 18 nodes per lane
  const int base   = (FPAD == 32) ? 0  : 18 * (lane >> 4);
  const int fl     = (FPAD == 32) ? lane : lrow;
  #pragma unroll
  for (int p = 0; p < npairs; ++p) {
    const int n0 = base + 2 * p;
    float a0, a1;
    if (FIN == 2) {
      float2 h0 = *(const float2*)(Hw + n0 * HS);
      float2 h1 = *(const float2*)(Hw + (n0 + 1) * HS);
      a0 = h0.x * Wc[0] + h0.y * Wc[1];
      a1 = h1.x * Wc[0] + h1.y * Wc[1];
    } else {
      a0 = 0.f; a1 = 0.f;
      const float4* hv0 = (const float4*)(Hw + n0 * HS);
      const float4* hv1 = (const float4*)(Hw + (n0 + 1) * HS);
      #pragma unroll
      for (int q = 0; q < FIN / 4; ++q) {
        float4 h0 = hv0[q], h1 = hv1[q];
        a0 += h0.x * Wc[4*q] + h0.y * Wc[4*q+1] + h0.z * Wc[4*q+2] + h0.w * Wc[4*q+3];
        a1 += h1.x * Wc[4*q] + h1.y * Wc[4*q+1] + h1.z * Wc[4*q+2] + h1.w * Wc[4*q+3];
      }
    }
    v2h pk = { (_Float16)a0, (_Float16)a1 };
    *(v2h*)(Tw + fl * 64 + n0) = pk;
  }
}

// Shared WMMA core: D[ct][i] = sum_k A_tile(i,k) x B_tile(k, ct).
// B operand: lane -> N = lrow (+16ct); elems 0..7 / 8..15 are 8 consecutive
// K(=node) values -> two ds_load_b128 from the zero-padded f16 T rows.
template <int NCT>
__device__ __forceinline__ void wmma_tiles(v8f (&D)[NCT][3], const _Float16* Tw,
                                           const v16h (&At)[3][2],
                                           int lrow, int khalf) {
  v8f zero = {};
  #pragma unroll
  for (int c = 0; c < NCT; ++c)
    #pragma unroll
    for (int i = 0; i < 3; ++i) D[c][i] = zero;
  #pragma unroll
  for (int ct = 0; ct < NCT; ++ct) {
    const int f = lrow + 16 * ct;
    #pragma unroll
    for (int k = 0; k < 2; ++k) {
      const int node0 = 32 * k + khalf;
      const v8h* p = (const v8h*)(Tw + f * 64 + node0);
      v8h lo = p[0];          // K = node0 .. node0+7
      v8h hi = p[2];          // K = node0+16 .. node0+23
      v16h Bt = __builtin_shufflevector(lo, hi,
                  0,1,2,3,4,5,6,7, 8,9,10,11,12,13,14,15);
      #pragma unroll
      for (int i = 0; i < 3; ++i)
        D[ct][i] = __builtin_amdgcn_wmma_f32_16x16x32_f16(
            false, At[i][k], false, Bt, (short)0, D[ct][i], false, false);
    }
  }
}

// Layers 0-2: H' = relu(A_hat @ T + b) written back to LDS (stride-16 rows).
// C/D layout: lane -> N = lrow, VGPR vv -> M = 16*i + vv + khalf; H padded to
// 48 rows -> unconditional stores; junk rows/cols are never read.
__device__ __forceinline__ void propagate(float* Hw, const _Float16* Tw,
                                          const v16h (&At)[3][2],
                                          float bias, int lrow, int khalf) {
  v8f D[1][3];
  wmma_tiles<1>(D, Tw, At, lrow, khalf);
  #pragma unroll
  for (int i = 0; i < 3; ++i)
    #pragma unroll
    for (int vv = 0; vv < 8; ++vv) {
      const int node = 16 * i + vv + khalf;
      Hw[node * HS + lrow] = fmaxf(D[0][i][vv] + bias, 0.f);
    }
}

// Layer 3 + pooling, entirely in registers: each feature column of D lives in
// lanes {L, L^16}; masked max/sum over valid nodes then one shfl_xor combine.
// Result: feature f = lane holds (gmp, gap) -> float2.
__device__ __forceinline__ float2 propagate_pool(const _Float16* Tw,
                                                 const v16h (&At)[3][2],
                                                 const float (&bv)[2],
                                                 int lane, int lrow, int khalf) {
  v8f D[2][3];
  wmma_tiles<2>(D, Tw, At, lrow, khalf);
  float mxc[2], smc[2];
  #pragma unroll
  for (int ct = 0; ct < 2; ++ct) {
    float mx = -3.4e38f, sm = 0.f;
    #pragma unroll
    for (int i = 0; i < 3; ++i)
      #pragma unroll
      for (int vv = 0; vv < 8; ++vv) {
        const int node = 16 * i + vv + khalf;
        const float v = fmaxf(D[ct][i][vv] + bv[ct], 0.f);
        const bool ok = node < NPG;
        mx = ok ? fmaxf(mx, v) : mx;
        sm = ok ? sm + v : sm;
      }
    mx = fmaxf(mx, __shfl_xor(mx, 16, 32));
    sm = sm + __shfl_xor(sm, 16, 32);
    mxc[ct] = mx; smc[ct] = sm;
  }
  const float mxf = (lane < 16) ? mxc[0] : mxc[1];   // feature f = lane
  const float smf = (lane < 16) ? smc[0] : smc[1];
  return make_float2(mxf, smf * (1.f / 36.f));
}

__global__ __launch_bounds__(128) void gcn_main(
    const float* __restrict__ x,
    const float* __restrict__ W0, const float* __restrict__ b0,
    const float* __restrict__ W1, const float* __restrict__ b1,
    const float* __restrict__ W2, const float* __restrict__ b2,
    const float* __restrict__ W3, const float* __restrict__ b3,
    const float* __restrict__ Wfc, const float* __restrict__ bfc,
    const float* __restrict__ Ahat,
    float* __restrict__ out, float* __restrict__ hid, int nGraphs)
{
  __shared__ float sA[48 * 64];                       // padded A_hat (block)
  __shared__ float sH[4][48 * HS];                    // per-wave H (layers 0-2)
  __shared__ __align__(16) _Float16 sTf[4][32 * 64];  // per-wave T, f16, [f][node]

  const int lane  = threadIdx.x & 31;
  const int wv    = threadIdx.x >> 5;
  const int lrow  = lane & 15;
  const int khalf = (lane < 16) ? 0 : 8;

  for (int i = threadIdx.x; i < 48 * 64; i += blockDim.x) sA[i] = Ahat[i];
  __syncthreads();

  // A operand tiles in the 16-bit A layout:
  // lane -> M = lrow; elem e -> K = 32k + (e<8 ? e : e+8) + khalf
  v16h At[3][2];
  #pragma unroll
  for (int i = 0; i < 3; ++i)
    #pragma unroll
    for (int k = 0; k < 2; ++k) {
      v16h a;
      #pragma unroll
      for (int e = 0; e < 16; ++e) {
        const int col = 32 * k + ((e < 8) ? e : e + 8) + khalf;
        a[e] = (_Float16)sA[(16 * i + lrow) * 64 + col];
      }
      At[i][k] = a;
    }

  float*    Hw = sH[wv];
  _Float16* Tw = sTf[wv];

  // Zero T once: node rows 36..63 stay zero forever (transform writes n<36).
  {
    unsigned int* z = (unsigned int*)Tw;
    for (int i = lane; i < 32 * 64 / 2; i += 32) z[i] = 0u;
  }

  // Hoist per-lane weight columns (zero-masked beyond fout) and biases.
  const int c4  = (lrow < 4)  ? lrow : 3;
  const int c12 = (lrow < 12) ? lrow : 11;
  float Wc0[2], Wc1[4], Wc2[12], Wc3[16];
  #pragma unroll
  for (int kk = 0; kk < 2;  ++kk) { float w = W0[kk*4  + c4];  Wc0[kk] = (lrow < 4)  ? w : 0.f; }
  #pragma unroll
  for (int kk = 0; kk < 4;  ++kk) { float w = W1[kk*12 + c12]; Wc1[kk] = (lrow < 12) ? w : 0.f; }
  #pragma unroll
  for (int kk = 0; kk < 12; ++kk) Wc2[kk] = W2[kk*16 + lrow];
  #pragma unroll
  for (int kk = 0; kk < 16; ++kk) Wc3[kk] = W3[kk*32 + lane];
  const float bi0 = b0[c4];               // clamped; junk columns never read
  const float bi1 = b1[c12];
  const float bi2 = b2[lrow];
  const float bv3[2] = { b3[lrow], b3[lrow + 16] };
  const int   jfc  = (lane < 7) ? lane : 6;    // clamped Wfc column
  const float bfcl = bfc[jfc];

  const int waveId = blockIdx.x * 4 + wv;
  const int nWaves = gridDim.x * 4;

  for (int g = waveId; g < nGraphs; g += nWaves) {
    // x[36,2] -> H (one float2 per node)
    for (int n = lane; n < NPG; n += 32)
      *(float2*)(Hw + n * HS) = *(const float2*)(x + ((size_t)g * NPG + n) * 2);

    transform<2, 16>(Hw, Tw, Wc0, lane, lrow);
    propagate(Hw, Tw, At, bi0, lrow, khalf);
    transform<4, 16>(Hw, Tw, Wc1, lane, lrow);
    propagate(Hw, Tw, At, bi1, lrow, khalf);
    transform<12, 16>(Hw, Tw, Wc2, lane, lrow);
    propagate(Hw, Tw, At, bi2, lrow, khalf);
    transform<16, 32>(Hw, Tw, Wc3, lane, lrow);
    const float2 pool = propagate_pool(Tw, At, bv3, lane, lrow, khalf);

    hid[(size_t)g * 64 + lane]      = pool.x;   // gmp
    hid[(size_t)g * 64 + 32 + lane] = pool.y;   // gap

    // FC 64->7 via wave shuffles: hidden[t]=shfl(gmp,t), hidden[t+32]=shfl(gap,t)
    float o = bfcl;
    #pragma unroll 8
    for (int t = 0; t < 32; ++t) {
      const float hm = __shfl(pool.x, t, 32);
      const float ha = __shfl(pool.y, t, 32);
      o += hm * Wfc[t * 7 + jfc] + ha * Wfc[(t + 32) * 7 + jfc];
    }
    if (lane < 7) out[(size_t)g * 7 + lane] = fmaxf(o, 0.f);
  }
}

// ---------------------------------------------------------------------------
extern "C" void kernel_launch(void* const* d_in, const int* in_sizes, int n_in,
                              void* d_out, int out_size, void* d_ws, size_t ws_size,
                              hipStream_t stream) {
  const float* x          = (const float*)d_in[0];
  const int*   edge_index = (const int*)d_in[1];
  // d_in[2] = pos (unused), d_in[3] = batch_index (implicit: 36 nodes/graph)
  const float* W0  = (const float*)d_in[4];
  const float* b0  = (const float*)d_in[5];
  const float* W1  = (const float*)d_in[6];
  const float* b1  = (const float*)d_in[7];
  const float* W2  = (const float*)d_in[8];
  const float* b2  = (const float*)d_in[9];
  const float* W3  = (const float*)d_in[10];
  const float* b3  = (const float*)d_in[11];
  const float* Wfc = (const float*)d_in[12];
  const float* bfc = (const float*)d_in[13];

  const int nGraphs = out_size / 71;            // out: B*7 + hidden: B*64
  const int nEdges  = in_sizes[1] / 2;          // total directed edges
  const int* srcRow = edge_index;               // first 220 = graph 0 (0..35)
  const int* dstRow = edge_index + nEdges;

  float* Aws = (float*)d_ws;                    // 48*64*4 = 12 KB of scratch
  gcn_prep<<<1, 64, 0, stream>>>(srcRow, dstRow, Aws);

  float* outp = (float*)d_out;
  float* hidp = outp + (size_t)nGraphs * 7;
  gcn_main<<<512, 128, 0, stream>>>(x, W0, b0, W1, b1, W2, b2, W3, b3,
                                    Wfc, bfc, Aws, outp, hidp, nGraphs);
}
